// GIN_28424093565719
// MI455X (gfx1250) — matrix operand order repaired and checked by reading, not compiled, not run on hardware
//
#include <hip/hip_runtime.h>

#define NNODES  100000
#define NEDGES  1200000
#define DIM     64
#define NCLASS  10
#define GIN_EPSF 0.5f
#define BN_EPSF  1e-3f

typedef __attribute__((ext_vector_type(16))) _Float16 v16h;
typedef __attribute__((ext_vector_type(8)))  float    v8f;

// ---------------------------------------------------------------------------
// agg init: out = (1 + eps) * in            (vectorized float4 sweep)
// ---------------------------------------------------------------------------
__global__ __launch_bounds__(256)
void scale_copy_kernel(const float* __restrict__ in, float* __restrict__ out, int n4) {
    int i = blockIdx.x * blockDim.x + threadIdx.x;
    if (i >= n4) return;
    float4 v = reinterpret_cast<const float4*>(in)[i];
    const float s = 1.0f + GIN_EPSF;
    v.x *= s; v.y *= s; v.z *= s; v.w *= s;
    reinterpret_cast<float4*>(out)[i] = v;
}

// ---------------------------------------------------------------------------
// edge scatter: agg[dst] += w * h[src]
// one thread per (edge, 4-feature chunk): float4 gather (L2-resident table),
// 4x global_atomic_add_f32 into the L2-resident accumulator.
// ---------------------------------------------------------------------------
__global__ __launch_bounds__(256)
void edge_scatter_kernel(const float* __restrict__ h,
                         const int*   __restrict__ src,
                         const int*   __restrict__ dst,
                         const float* __restrict__ ew,
                         float*       __restrict__ agg) {
    int idx = blockIdx.x * blockDim.x + threadIdx.x;   // E*16 = 19.2M < 2^31
    int e = idx >> 4;
    if (e >= NEDGES) return;
    int c = (idx & 15) << 2;
    int s = src[e];
    int d = dst[e];
    float w = ew[e];
    float4 v = *reinterpret_cast<const float4*>(h + (size_t)s * DIM + c);
    float* base = agg + (size_t)d * DIM + c;
    atomicAdd(base + 0, w * v.x);
    atomicAdd(base + 1, w * v.y);
    atomicAdd(base + 2, w * v.z);
    atomicAdd(base + 3, w * v.w);
}

// ---------------------------------------------------------------------------
// Fused MLP GEMM:  out = epilogue( A[nrows x 64] @ W[64 x ncol] + bias )
//   epilogue: optional BN (inference affine), optional ReLU.
// Block = 256 threads = 8 waves; each wave computes a 16-row x (NCT*16)-col
// stripe via v_wmma_f32_16x16x32_f16 (K=64 -> 2 k-steps per column tile).
// W is converted to f16 and held transposed in LDS (stride 80 halves so each
// v16h B-fragment read is a 32B-aligned contiguous ds_load).
// ---------------------------------------------------------------------------
template <bool BN, bool RELU, int NCT>
__global__ __launch_bounds__(256)
void mlp_gemm_kernel(const float* __restrict__ A,
                     const float* __restrict__ W,     // [64 x ncol] row-major
                     const float* __restrict__ bias,  // [ncol]
                     const float* __restrict__ gamma,
                     const float* __restrict__ beta,
                     const float* __restrict__ mean,
                     const float* __restrict__ var,
                     float* __restrict__ out,         // [nrows x ncol]
                     int nrows, int ncol) {
    constexpr int NCOLS = NCT * 16;
    constexpr int WSTR  = DIM + 16;                  // 80 halves = 160B row stride
    __shared__ _Float16 WT[NCOLS * WSTR];

    const int tid = threadIdx.x;

    // Cooperative load: WT[n][k] = (f16) W[k][n], zero-padded columns.
    for (int i = tid; i < DIM * NCOLS; i += 256) {
        int k = i / NCOLS;
        int n = i - k * NCOLS;
        float v = (n < ncol) ? W[k * ncol + n] : 0.0f;
        WT[n * WSTR + k] = (_Float16)v;
    }
    __syncthreads();

    const int wave = tid >> 5;
    const int lane = tid & 31;
    const int ln16 = lane & 15;
    const int hi   = lane >> 4;                      // lane group: 0 or 1
    const int rbase = (blockIdx.x * 8 + wave) * 16;  // first row of this wave's tile

    // ---- A fragments (16x32 f16 layout), both K-steps, rows clamped so EXEC
    //      stays all-ones for WMMA; out-of-range rows are dropped at store.
    int arow = rbase + ln16;
    if (arow > nrows - 1) arow = nrows - 1;
    const float* ap = A + (size_t)arow * DIM + hi * 16;

    v16h afr[2];
#pragma unroll
    for (int s = 0; s < 2; ++s) {
        const float* p = ap + s * 32;
#pragma unroll
        for (int j = 0; j < 8; ++j) afr[s][j]     = (_Float16)p[j];        // K = kb+0..7
#pragma unroll
        for (int j = 0; j < 8; ++j) afr[s][8 + j] = (_Float16)p[8 + j];    // K = kb+8..15
    }

    // ---- accumulate
    v8f acc[NCT];
#pragma unroll
    for (int t = 0; t < NCT; ++t)
#pragma unroll
        for (int v = 0; v < 8; ++v) acc[t][v] = 0.0f;

#pragma unroll
    for (int t = 0; t < NCT; ++t) {
        const int col = t * 16 + ln16;               // B-frag column for this lane
#pragma unroll
        for (int s = 0; s < 2; ++s) {
            const int klo = s * 32 + hi * 16;        // B-frag K-run for this lane
            v16h bfr = *reinterpret_cast<const v16h*>(&WT[col * WSTR + klo]);
            acc[t] = __builtin_amdgcn_wmma_f32_16x16x32_f16(
                false, afr[s], false, bfr, (short)0, acc[t], false, false);
        }
    }

    // ---- fused epilogue: bias (+BN) (+ReLU), store f32
#pragma unroll
    for (int t = 0; t < NCT; ++t) {
        const int col = t * 16 + ln16;
        if (col >= ncol) continue;
        float b  = bias[col];
        float sc = 1.0f, sh = 0.0f;
        if (BN) {
            float rs = rsqrtf(var[col] + BN_EPSF) * gamma[col];
            sc = rs;
            sh = beta[col] - mean[col] * rs;
        }
#pragma unroll
        for (int v = 0; v < 8; ++v) {
            int row = rbase + v + hi * 8;            // C/D layout: VGPR v, lane group hi
            if (row >= nrows) continue;
            float x = acc[t][v] + b;
            if (BN)   x = x * sc + sh;
            if (RELU) x = fmaxf(x, 0.0f);
            out[(size_t)row * ncol + col] = x;
        }
    }
}

// ---------------------------------------------------------------------------
extern "C" void kernel_launch(void* const* d_in, const int* in_sizes, int n_in,
                              void* d_out, int out_size, void* d_ws, size_t ws_size,
                              hipStream_t stream) {
    const float* x   = (const float*)d_in[0];
    const int*   ei  = (const int*)  d_in[1];
    const float* ew  = (const float*)d_in[2];
    const float* W1a = (const float*)d_in[3];
    const float* b1a = (const float*)d_in[4];
    const float* W1b = (const float*)d_in[5];
    const float* b1b = (const float*)d_in[6];
    const float* g1  = (const float*)d_in[7];
    const float* be1 = (const float*)d_in[8];
    const float* m1  = (const float*)d_in[9];
    const float* v1  = (const float*)d_in[10];
    const float* W2a = (const float*)d_in[11];
    const float* b2a = (const float*)d_in[12];
    const float* W2b = (const float*)d_in[13];
    const float* b2b = (const float*)d_in[14];
    const float* g2  = (const float*)d_in[15];
    const float* be2 = (const float*)d_in[16];
    const float* m2  = (const float*)d_in[17];
    const float* v2  = (const float*)d_in[18];

    const int* src = ei;             // edge_index row 0
    const int* dst = ei + NEDGES;    // edge_index row 1
    float* out = (float*)d_out;

    float* bufA = (float*)d_ws;                       // N x 64
    float* bufB = bufA + (size_t)NNODES * DIM;        // N x 64

    const dim3 blk(256);
    const int  n4 = NNODES * DIM / 4;
    const dim3 gScale((n4 + 255) / 256);
    const dim3 gEdge((NEDGES * 16 + 255) / 256);
    const dim3 gGemm((((NNODES + 15) / 16) + 7) / 8);

    // Layer 1: agg -> Dense+ReLU -> Dense+BN+ReLU
    scale_copy_kernel <<<gScale, blk, 0, stream>>>(x, bufA, n4);
    edge_scatter_kernel<<<gEdge, blk, 0, stream>>>(x, src, dst, ew, bufA);
    mlp_gemm_kernel<false, true, 4><<<gGemm, blk, 0, stream>>>(
        bufA, W1a, b1a, nullptr, nullptr, nullptr, nullptr, bufB, NNODES, DIM);
    mlp_gemm_kernel<true,  true, 4><<<gGemm, blk, 0, stream>>>(
        bufB, W1b, b1b, g1, be1, m1, v1, bufA, NNODES, DIM);

    // Layer 2: agg -> Dense+ReLU -> Dense+BN
    scale_copy_kernel <<<gScale, blk, 0, stream>>>(bufA, bufB, n4);
    edge_scatter_kernel<<<gEdge, blk, 0, stream>>>(bufA, src, dst, ew, bufB);
    mlp_gemm_kernel<false, true, 4><<<gGemm, blk, 0, stream>>>(
        bufB, W2a, b2a, nullptr, nullptr, nullptr, nullptr, bufA, NNODES, DIM);
    mlp_gemm_kernel<true, false, 1><<<gGemm, blk, 0, stream>>>(
        bufA, W2b, b2b, g2, be2, m2, v2, out, NNODES, NCLASS);
}
